// QueryCrossAttention_88819923681568
// MI455X (gfx1250) — matrix operand order repaired and checked by reading, not compiled, not run on hardware
//
#include <hip/hip_runtime.h>
#include <hip/hip_bf16.h>

typedef _Float16 f16;
typedef __attribute__((ext_vector_type(16))) _Float16 v16h;
typedef __attribute__((ext_vector_type(8)))  _Float16 v8h;
typedef __attribute__((ext_vector_type(8)))  float    v8f;

static __device__ inline v8f zero8() {
    v8f z = {0.f, 0.f, 0.f, 0.f, 0.f, 0.f, 0.f, 0.f};
    return z;
}

// Assemble a WMMA v16h fragment from two 16-byte (8-half) chunks.
static __device__ inline v16h ld_frag(const f16* p0, const f16* p1) {
    union { v16h v; v8h h[2]; } u;
    u.h[0] = *(const v8h*)p0;
    u.h[1] = *(const v8h*)p1;
    return u.v;
}

static __device__ inline v8f wmma16(const v16h& a, const v16h& b, const v8f& c) {
    return __builtin_amdgcn_wmma_f32_16x16x32_f16(false, a, false, b, (short)0, c,
                                                  false, false);
}

// LDS byte offset of a generic pointer to __shared__ memory: the generic
// pointer is {SHARED_BASE[63:32], lds_offset[31:0]} (ISA 10.2 aperture rule).
static __device__ inline unsigned int lds_off(const void* p) {
    return (unsigned int)(unsigned long long)p;
}

// CDNA5 async copy engine: global -> LDS, 16B per lane, tracked by ASYNCcnt.
static __device__ inline void async_load_b128(unsigned int lds, const f16* g) {
    asm volatile("global_load_async_to_lds_b128 %0, %1, off"
                 :: "v"(lds), "v"(g) : "memory");
}
static __device__ inline void wait_async() {
    asm volatile("s_wait_asynccnt 0" ::: "memory");
}

// ---------------------------------------------------------------------------
// Elementwise f32 -> f16 convert
// ---------------------------------------------------------------------------
__global__ void cvt_f16_kernel(const float* __restrict__ in, f16* __restrict__ out, int n) {
    int i = blockIdx.x * blockDim.x + threadIdx.x;
    int stride = gridDim.x * blockDim.x;
    for (; i < n; i += stride) out[i] = (f16)in[i];
}

// f32 (R x C) -> f16 transposed (C x R)
__global__ void cvt_f16_tr_kernel(const float* __restrict__ in, f16* __restrict__ out,
                                  int R, int C) {
    int i = blockIdx.x * blockDim.x + threadIdx.x;
    int n = R * C;
    int stride = gridDim.x * blockDim.x;
    for (; i < n; i += stride) {
        int r = i / C, c = i - r * C;
        out[(size_t)c * R + r] = (f16)in[i];
    }
}

// ---------------------------------------------------------------------------
// Tiled WMMA GEMM:  C(MxN) = A(MxK, row-major f16) * BT(NxK, row-major f16)^T
// Block: 256 threads (8 waves). Block tile 256x128; wave tile 64x64.
// Double-buffered LDS filled by the async copy engine (ASYNCcnt).
// mode 0: outF[m*N+n] = acc + bias[n]                       (f32 out)
// mode 1: qp scatter:  qph[b][h][nq][hd] = acc * 0.125      (f16)
// mode 2: kv scatter:  n<1024 -> kh[b][h][nkv][hd]
//                      n>=1024 -> vT[b][h][hd][nkv]         (f16)
// ---------------------------------------------------------------------------
__global__ __launch_bounds__(256) void gemm_f16_kernel(
    const f16* __restrict__ A, const f16* __restrict__ BT,
    int M, int N, int K, int mode,
    float* __restrict__ outF, f16* __restrict__ outP, f16* __restrict__ outV,
    const float* __restrict__ bias)
{
    __shared__ f16 Ash[2][256 * 40];
    __shared__ f16 Bsh[2][128 * 40];

    const int tid = threadIdx.x;
    const int w = tid >> 5, l = tid & 31;
    const int waveM = w >> 1;          // 0..3  -> 64 rows each
    const int waveN = w & 1;           // 0..1  -> 64 cols each
    const int lr = l & 15, hw = l >> 4;
    const int bm = blockIdx.y * 256, bn = blockIdx.x * 128;

    v8f acc[4][4];
#pragma unroll
    for (int mt = 0; mt < 4; ++mt)
#pragma unroll
        for (int nt = 0; nt < 4; ++nt) acc[mt][nt] = zero8();

    // async staging of one 256x32 A tile + 128x32 B tile into buffer `sel`
    auto stage = [&](int k0, int sel) {
        const f16* ga = A + (size_t)(bm + tid) * K + k0;          // row = tid
        unsigned int la = lds_off(&Ash[sel][tid * 40]);
#pragma unroll
        for (int c = 0; c < 32; c += 8)
            async_load_b128(la + c * 2, ga + c);
        const int brow = tid >> 1, bcol = (tid & 1) * 16;
        const f16* gb = BT + (size_t)(bn + brow) * K + k0 + bcol;
        unsigned int lb = lds_off(&Bsh[sel][brow * 40 + bcol]);
        async_load_b128(lb, gb);
        async_load_b128(lb + 16, gb + 8);
    };

    int sel = 0;
    stage(0, 0);
    wait_async();
    __syncthreads();

    for (int k0 = 0; k0 < K; k0 += 32) {
        if (k0 + 32 < K) stage(k0 + 32, sel ^ 1);

        v16h afr[4], bfr[4];
#pragma unroll
        for (int mt = 0; mt < 4; ++mt) {
            int row = waveM * 64 + mt * 16 + lr;
            int o = row * 40 + hw * 8;
            afr[mt] = ld_frag(&Ash[sel][o], &Ash[sel][o + 16]);
        }
#pragma unroll
        for (int nt = 0; nt < 4; ++nt) {
            int row = waveN * 64 + nt * 16 + lr;
            int o = row * 40 + hw * 8;
            bfr[nt] = ld_frag(&Bsh[sel][o], &Bsh[sel][o + 16]);
        }
#pragma unroll
        for (int mt = 0; mt < 4; ++mt)
#pragma unroll
            for (int nt = 0; nt < 4; ++nt)
                acc[mt][nt] = wmma16(afr[mt], bfr[nt], acc[mt][nt]);

        wait_async();
        __syncthreads();
        sel ^= 1;
    }

    // epilogue: element (m, n) lives in acc[mt][nt][r] at lane (hw, lr)
#pragma unroll
    for (int mt = 0; mt < 4; ++mt) {
#pragma unroll
        for (int nt = 0; nt < 4; ++nt) {
            const int n = bn + waveN * 64 + nt * 16 + lr;
#pragma unroll
            for (int r = 0; r < 8; ++r) {
                const int m = bm + waveM * 64 + mt * 16 + hw * 8 + r;
                float v = acc[mt][nt][r];
                if (mode == 0) {
                    outF[(size_t)m * N + n] = v + bias[n];
                } else if (mode == 1) {
                    // m = b*1024 + nq ; n = h*64 + d
                    int b = m >> 10, nq = m & 1023;
                    int h = n >> 6, d = n & 63;
                    outP[(((size_t)(b * 16 + h)) * 1024 + nq) * 64 + d] =
                        (f16)(v * 0.125f);
                } else {
                    // m = b*4096 + nkv ; n in [0,2048)
                    int b = m >> 12, nkv = m & 4095;
                    if (n < 1024) {
                        int h = n >> 6, d = n & 63;
                        outP[(((size_t)(b * 16 + h)) * 4096 + nkv) * 64 + d] = (f16)v;
                    } else {
                        int n2 = n - 1024;
                        int h = n2 >> 6, d = n2 & 63;
                        outV[(((size_t)(b * 16 + h)) * 64 + d) * 4096 + nkv] = (f16)v;
                    }
                }
            }
        }
    }
}

// ---------------------------------------------------------------------------
// Flash attention:  per (b,h): softmax(Q K^T + mask) V
// qph: [B,H,1024,64] f16 (pre-scaled)  kh: [B,H,4096,64]  vT: [B,H,64,4096]
// Block 256 = 8 waves; wave owns 16 q rows (block covers 128); kv chunk 64.
// K/V^T chunks double-buffered in LDS via async copy engine.
// Output xh: [B,1024,1024] f16 (heads re-interleaved for the proj GEMM).
// ---------------------------------------------------------------------------
__global__ __launch_bounds__(256) void flash_attn_kernel(
    const f16* __restrict__ qph, const f16* __restrict__ kh,
    const f16* __restrict__ vT, const unsigned char* __restrict__ mask,
    f16* __restrict__ xh)
{
    __shared__ f16 Ksh[2][64 * 72];
    __shared__ f16 Vsh[2][64 * 72];
    __shared__ f16 Psh[8 * 16 * 72];

    const int tid = threadIdx.x;
    const int w = tid >> 5, l = tid & 31;
    const int lr = l & 15, hw = l >> 4;
    const int bh = blockIdx.y;
    const int b = bh >> 4, h = bh & 15;
    const int q0 = blockIdx.x * 128 + w * 16;

    // Q fragments stay in registers for the whole kv loop
    const f16* qrow = qph + ((size_t)bh * 1024 + q0 + lr) * 64;
    v16h qf[2];
    qf[0] = ld_frag(qrow + hw * 8,      qrow + hw * 8 + 16);
    qf[1] = ld_frag(qrow + 32 + hw * 8, qrow + 32 + hw * 8 + 16);

    v8f o[4];
#pragma unroll
    for (int ht = 0; ht < 4; ++ht) o[ht] = zero8();
    float mi[8], li[8];
#pragma unroll
    for (int r = 0; r < 8; ++r) { mi[r] = -3.0e38f; li[r] = 0.f; }

    const int srow = tid >> 2;            // 0..63
    const int scol = (tid & 3) * 16;      // 0/16/32/48
    f16* pb = &Psh[w * 16 * 72];          // per-wave P tile

    auto stageKV = [&](int kv0, int sel) {
        const f16* gk = kh + ((size_t)bh * 4096 + kv0 + srow) * 64 + scol;
        const f16* gv = vT + ((size_t)bh * 64 + srow) * 4096 + kv0 + scol;
        unsigned int lk = lds_off(&Ksh[sel][srow * 72 + scol]);
        unsigned int lv = lds_off(&Vsh[sel][srow * 72 + scol]);
        async_load_b128(lk,      gk);
        async_load_b128(lk + 16, gk + 8);
        async_load_b128(lv,      gv);
        async_load_b128(lv + 16, gv + 8);
    };

    int sel = 0;
    stageKV(0, 0);
    wait_async();
    __syncthreads();

    for (int kv0 = 0; kv0 < 4096; kv0 += 64) {
        if (kv0 + 64 < 4096) stageKV(kv0 + 64, sel ^ 1);

        // S = Q K^T  (16 x 64 per wave)
        v8f s[4];
#pragma unroll
        for (int nt = 0; nt < 4; ++nt) {
            int row = nt * 16 + lr;
            v16h b0 = ld_frag(&Ksh[sel][row * 72 + hw * 8],
                              &Ksh[sel][row * 72 + hw * 8 + 16]);
            v16h b1 = ld_frag(&Ksh[sel][row * 72 + 32 + hw * 8],
                              &Ksh[sel][row * 72 + 32 + hw * 8 + 16]);
            v8f t = zero8();
            t = wmma16(qf[0], b0, t);
            t = wmma16(qf[1], b1, t);
            s[nt] = t;
        }

        // key-padding mask (column-wise)
#pragma unroll
        for (int nt = 0; nt < 4; ++nt) {
            int kvc = kv0 + nt * 16 + lr;
            if (mask[(size_t)b * 4096 + kvc]) {
#pragma unroll
                for (int r = 0; r < 8; ++r) s[nt][r] = -1.0e30f;
            }
        }

        // online softmax (rows striped over 16-lane halves)
        float al[8];
#pragma unroll
        for (int r = 0; r < 8; ++r) {
            float cm = s[0][r];
#pragma unroll
            for (int nt = 1; nt < 4; ++nt) cm = fmaxf(cm, s[nt][r]);
#pragma unroll
            for (int off = 8; off >= 1; off >>= 1)
                cm = fmaxf(cm, __shfl_xor(cm, off, 32));
            float mn = fmaxf(mi[r], cm);
            al[r] = __expf(mi[r] - mn);
            float sum = 0.f;
#pragma unroll
            for (int nt = 0; nt < 4; ++nt) {
                float e = __expf(s[nt][r] - mn);
                s[nt][r] = e;
                sum += e;
            }
#pragma unroll
            for (int off = 8; off >= 1; off >>= 1)
                sum += __shfl_xor(sum, off, 32);
            li[r] = li[r] * al[r] + sum;
            mi[r] = mn;
        }
#pragma unroll
        for (int ht = 0; ht < 4; ++ht)
#pragma unroll
            for (int r = 0; r < 8; ++r) o[ht][r] *= al[r];

        // re-shard P into WMMA A layout via per-wave LDS tile
        // (per-wave DS ops are in-order on CDNA5, no barrier needed)
#pragma unroll
        for (int nt = 0; nt < 4; ++nt)
#pragma unroll
            for (int r = 0; r < 8; ++r)
                pb[(hw * 8 + r) * 72 + nt * 16 + lr] = (f16)s[nt][r];

        const f16* pr = pb + lr * 72;
        v16h pf[2];
        pf[0] = ld_frag(pr + hw * 8,      pr + hw * 8 + 16);
        pf[1] = ld_frag(pr + 32 + hw * 8, pr + 32 + hw * 8 + 16);

        // O += P V
#pragma unroll
        for (int ht = 0; ht < 4; ++ht) {
            int row = ht * 16 + lr;
            v16h v0 = ld_frag(&Vsh[sel][row * 72 + hw * 8],
                              &Vsh[sel][row * 72 + hw * 8 + 16]);
            v16h v1 = ld_frag(&Vsh[sel][row * 72 + 32 + hw * 8],
                              &Vsh[sel][row * 72 + 32 + hw * 8 + 16]);
            o[ht] = wmma16(pf[0], v0, o[ht]);
            o[ht] = wmma16(pf[1], v1, o[ht]);
        }

        wait_async();
        __syncthreads();
        sel ^= 1;
    }

    // normalize and write out (re-interleave heads: [b, nq, h*64+d])
#pragma unroll
    for (int r = 0; r < 8; ++r) {
        float inv = 1.0f / li[r];
        int m = q0 + hw * 8 + r;
#pragma unroll
        for (int ht = 0; ht < 4; ++ht) {
            int col = h * 64 + ht * 16 + lr;
            xh[((size_t)b * 1024 + m) * 1024 + col] = (f16)(o[ht][r] * inv);
        }
    }
}

// ---------------------------------------------------------------------------
// Host-side orchestration
// ---------------------------------------------------------------------------
extern "C" void kernel_launch(void* const* d_in, const int* in_sizes, int n_in,
                              void* d_out, int out_size, void* d_ws, size_t ws_size,
                              hipStream_t stream) {
    const float* q     = (const float*)d_in[0];   // (4,1024,1024)
    const float* kv    = (const float*)d_in[1];   // (4,4096,1024)
    const unsigned char* mask = (const unsigned char*)d_in[2];  // (4,4096) bool
    const float* Wq    = (const float*)d_in[3];   // (1024,1024)
    const float* Wkv   = (const float*)d_in[4];   // (1024,2048)
    const float* Wproj = (const float*)d_in[5];   // (1024,1024)
    const float* bproj = (const float*)d_in[6];   // (1024,)

    const size_t MB = 1ull << 20;
    char* ws = (char*)d_ws;
    f16* qh     = (f16*)(ws + 0 * MB);     // 4M  f16 ( 8MB)
    f16* kvh    = (f16*)(ws + 8 * MB);     // 16M f16 (32MB)
    f16* WqT    = (f16*)(ws + 40 * MB);    // 1M  f16 ( 2MB)
    f16* WkvT   = (f16*)(ws + 42 * MB);    // 2M  f16 ( 4MB)
    f16* WprojT = (f16*)(ws + 46 * MB);    // 1M  f16 ( 2MB)
    f16* qph    = (f16*)(ws + 48 * MB);    // 4M  f16 ( 8MB)
    f16* kH     = (f16*)(ws + 56 * MB);    // 16M f16 (32MB)
    f16* vT     = (f16*)(ws + 88 * MB);    // 16M f16 (32MB)
    f16* xh     = (f16*)(ws + 120 * MB);   // 4M  f16 ( 8MB)

    // 1) f32 -> f16 conversions / weight transposes
    cvt_f16_kernel<<<2048, 256, 0, stream>>>(q, qh, 4 * 1024 * 1024);
    cvt_f16_kernel<<<4096, 256, 0, stream>>>(kv, kvh, 16 * 1024 * 1024);
    cvt_f16_tr_kernel<<<2048, 256, 0, stream>>>(Wq, WqT, 1024, 1024);
    cvt_f16_tr_kernel<<<4096, 256, 0, stream>>>(Wkv, WkvT, 1024, 2048);
    cvt_f16_tr_kernel<<<2048, 256, 0, stream>>>(Wproj, WprojT, 1024, 1024);

    // 2) Q projection (scaled, scattered per-head)
    gemm_f16_kernel<<<dim3(8, 16), 256, 0, stream>>>(
        qh, WqT, 4096, 1024, 1024, 1, nullptr, qph, nullptr, nullptr);

    // 3) KV projection (K row-major per head, V transposed per head)
    gemm_f16_kernel<<<dim3(16, 64), 256, 0, stream>>>(
        kvh, WkvT, 16384, 2048, 1024, 2, nullptr, kH, vT, nullptr);

    // 4) flash attention over all (b,h)
    flash_attn_kernel<<<dim3(8, 64), 256, 0, stream>>>(qph, kH, vT, mask, xh);

    // 5) output projection + bias -> f32
    gemm_f16_kernel<<<dim3(8, 16), 256, 0, stream>>>(
        xh, WprojT, 4096, 1024, 1024, 0, (float*)d_out, nullptr, nullptr, bproj);
}
